// Skel_Hausdorff_Loss_26018911879199
// MI455X (gfx1250) — compile-verified, matrix operand order checked by reading, not run on hardware
//
#include <hip/hip_runtime.h>

// ---------------------------------------------------------------------------
// Skel+Hausdorff loss for MI455X (gfx1250, wave32).
//  - soft-skeleton: fused erode/erode/maxpool/update stencil, LDS tiles
//  - reductions: v_wmma_f32_16x16x32_f16 with B=ones (row-sum trick)
//  - EDT row pass: global_load_async_to_lds_b128 + s_wait_asynccnt
// Workspace layout (floats): A0..A5 = 6 x NPIX, then ACC[64].
// ---------------------------------------------------------------------------

#define HH   512
#define WW   512
#define NSL  8                       // B*C slices
#define NPIX (NSL * HH * WW)         // 2,097,152
#define TILE 32

typedef __attribute__((ext_vector_type(16))) _Float16 v16h;
typedef __attribute__((ext_vector_type(8)))  float    v8f;

__device__ __forceinline__ float crossMin(const float* s, int pitch, int idx) {
    float v = fminf(s[idx], fminf(s[idx - pitch], s[idx + pitch]));
    return fminf(v, fminf(s[idx - 1], s[idx + 1]));
}

// -------------------- accumulator helpers ----------------------------------
// ACC[0..3]  : cldice sums (skelP*yT, skelP, skelT*yP, skelT)
// ACC[8+s]   : sum d_p * ge   (g2p numerator)    ACC[16+s]: sum d_g * pe
// ACC[24+s]  : count ge       ACC[32+s]: count pe

__global__ void zero_acc_kernel(float* acc) {
    if (threadIdx.x < 64) acc[threadIdx.x] = 0.0f;
}

// -------------------- phase 1: soft skeleton -------------------------------

// sk = relu(img - maxpool3x3(erode(img)))  ; z<8 -> pred, z>=8 -> true
__global__ void skel_init_kernel(const float* __restrict__ imgP, float* __restrict__ skP,
                                 const float* __restrict__ imgT, float* __restrict__ skT) {
    const int z = blockIdx.z, sl = z & 7;
    const float* img = (z < 8 ? imgP : imgT) + (size_t)sl * HH * WW;
    float*       sk  = (z < 8 ? skP  : skT)  + (size_t)sl * HH * WW;
    const int bx = blockIdx.x * TILE, by = blockIdx.y * TILE;
    const int tid = threadIdx.x;
    const float FINF = __builtin_inff();

    __shared__ float sIm[36 * 36];
    __shared__ float sE1[34 * 34];

    for (int i = tid; i < 36 * 36; i += 256) {
        int ly = i / 36, lx = i % 36, gy = by + ly - 2, gx = bx + lx - 2;
        sIm[i] = (gy >= 0 && gy < HH && gx >= 0 && gx < WW) ? img[gy * WW + gx] : FINF;
    }
    __syncthreads();
    for (int i = tid; i < 34 * 34; i += 256) {
        int ly = i / 34, lx = i % 34, gy = by + ly - 1, gx = bx + lx - 1;
        float v = crossMin(sIm, 36, (ly + 1) * 36 + (lx + 1));
        if (!(gy >= 0 && gy < HH && gx >= 0 && gx < WW)) v = -FINF;  // ignored by max
        sE1[i] = v;
    }
    __syncthreads();
    for (int i = tid; i < TILE * TILE; i += 256) {
        int ly = i / TILE, lx = i % TILE;
        float o = -FINF;
        #pragma unroll
        for (int dy = 0; dy < 3; dy++)
            #pragma unroll
            for (int dx = 0; dx < 3; dx++)
                o = fmaxf(o, sE1[(ly + dy) * 34 + (lx + dx)]);
        float c = sIm[(ly + 2) * 36 + (lx + 2)];
        sk[(size_t)(by + ly) * WW + bx + lx] = fmaxf(c - o, 0.0f);
    }
}

// im' = erode(im); delta = relu(im' - maxpool3x3(erode(im'))); sk += relu(delta - sk*delta)
__global__ void skel_iter_kernel(const float* __restrict__ inP, float* __restrict__ outP,
                                 float* __restrict__ skP,
                                 const float* __restrict__ inT, float* __restrict__ outT,
                                 float* __restrict__ skT) {
    const int z = blockIdx.z, sl = z & 7;
    const float* im  = (z < 8 ? inP  : inT)  + (size_t)sl * HH * WW;
    float*       imo = (z < 8 ? outP : outT) + (size_t)sl * HH * WW;
    float*       sk  = (z < 8 ? skP  : skT)  + (size_t)sl * HH * WW;
    const int bx = blockIdx.x * TILE, by = blockIdx.y * TILE;
    const int tid = threadIdx.x;
    const float FINF = __builtin_inff();

    __shared__ float sIm[38 * 38];
    __shared__ float sE1[36 * 36];
    __shared__ float sE2[34 * 34];

    for (int i = tid; i < 38 * 38; i += 256) {
        int ly = i / 38, lx = i % 38, gy = by + ly - 3, gx = bx + lx - 3;
        sIm[i] = (gy >= 0 && gy < HH && gx >= 0 && gx < WW) ? im[gy * WW + gx] : FINF;
    }
    __syncthreads();
    for (int i = tid; i < 36 * 36; i += 256) {
        int ly = i / 36, lx = i % 36, gy = by + ly - 2, gx = bx + lx - 2;
        float v = crossMin(sIm, 38, (ly + 1) * 38 + (lx + 1));
        if (!(gy >= 0 && gy < HH && gx >= 0 && gx < WW)) v = FINF;   // ignored by next min
        sE1[i] = v;
    }
    __syncthreads();
    for (int i = tid; i < 34 * 34; i += 256) {
        int ly = i / 34, lx = i % 34, gy = by + ly - 1, gx = bx + lx - 1;
        float v = crossMin(sE1, 36, (ly + 1) * 36 + (lx + 1));
        if (!(gy >= 0 && gy < HH && gx >= 0 && gx < WW)) v = -FINF;  // ignored by max
        sE2[i] = v;
    }
    __syncthreads();
    for (int i = tid; i < TILE * TILE; i += 256) {
        int ly = i / TILE, lx = i % TILE;
        float o = -FINF;
        #pragma unroll
        for (int dy = 0; dy < 3; dy++)
            #pragma unroll
            for (int dx = 0; dx < 3; dx++)
                o = fmaxf(o, sE2[(ly + dy) * 34 + (lx + dx)]);
        float e1 = sE1[(ly + 2) * 36 + (lx + 2)];
        float delta = fmaxf(e1 - o, 0.0f);
        size_t gi = (size_t)(by + ly) * WW + bx + lx;
        float s = sk[gi];
        sk[gi]  = s + fmaxf(delta - s * delta, 0.0f);
        imo[gi] = e1;
    }
}

// -------------------- WMMA reduction: sum(skel*other), sum(skel) -----------
// B = ones  ->  every D column = row-sum of A; sum of all 256 D elems = 16*total.
__global__ void skel_reduce_kernel(const float* __restrict__ skel,
                                   const float* __restrict__ other,
                                   float* __restrict__ acc, int accIdx) {
    const int lane   = threadIdx.x & 31;
    const int wave   = (blockIdx.x * blockDim.x + threadIdx.x) >> 5;
    const int nWaves = (gridDim.x * blockDim.x) >> 5;

    v16h ones;
    #pragma unroll
    for (int i = 0; i < 16; i++) ones[i] = (_Float16)1.0f;
    v8f accP = {};   // sum of products
    v8f accS = {};   // sum of skel

    const int nChunks = NPIX / 512;
    for (int c = wave; c < nChunks; c += nWaves) {
        const float* ps = skel  + (size_t)c * 512 + lane * 16;
        const float* po = other + (size_t)c * 512 + lane * 16;
        __builtin_prefetch(ps + (size_t)nWaves * 512, 0, 1);   // global_prefetch_b8
        v16h aP, aS;
        #pragma unroll
        for (int i = 0; i < 16; i += 4) {
            float4 s4 = *(const float4*)(ps + i);
            float4 o4 = *(const float4*)(po + i);
            aS[i + 0] = (_Float16)s4.x; aP[i + 0] = (_Float16)(s4.x * o4.x);
            aS[i + 1] = (_Float16)s4.y; aP[i + 1] = (_Float16)(s4.y * o4.y);
            aS[i + 2] = (_Float16)s4.z; aP[i + 2] = (_Float16)(s4.z * o4.z);
            aS[i + 3] = (_Float16)s4.w; aP[i + 3] = (_Float16)(s4.w * o4.w);
        }
        accP = __builtin_amdgcn_wmma_f32_16x16x32_f16(false, aP, false, ones,
                                                      (short)0, accP, false, false);
        accS = __builtin_amdgcn_wmma_f32_16x16x32_f16(false, aS, false, ones,
                                                      (short)0, accS, false, false);
    }
    float sP = 0.0f, sS = 0.0f;
    #pragma unroll
    for (int i = 0; i < 8; i++) { sP += accP[i]; sS += accS[i]; }
    for (int off = 16; off; off >>= 1) {
        sP += __shfl_xor(sP, off, 32);
        sS += __shfl_xor(sS, off, 32);
    }
    if (lane == 0) {
        atomicAdd(&acc[accIdx + 0], sP * 0.0625f);   // /16
        atomicAdd(&acc[accIdx + 1], sS * 0.0625f);
    }
}

// -------------------- phase 2: edges + exact EDT ---------------------------

__global__ void edges_kernel(const float* __restrict__ yT, const float* __restrict__ yP,
                             float* __restrict__ ge, float* __restrict__ pe,
                             float* __restrict__ acc) {
    const int z = blockIdx.z, sl = z & 7;
    const bool isPred = (z >= 8);
    const float* in  = (isPred ? yP : yT) + (size_t)sl * HH * WW;
    float*       out = (isPred ? pe : ge) + (size_t)sl * HH * WW;
    const int accIdx = (isPred ? 32 : 24) + sl;
    const int bx = blockIdx.x * TILE, by = blockIdx.y * TILE;
    const int tid = threadIdx.x;
    const float FINF = __builtin_inff();

    __shared__ float sM[34 * 34];
    __shared__ float red[256];

    for (int i = tid; i < 34 * 34; i += 256) {
        int ly = i / 34, lx = i % 34, gy = by + ly - 1, gx = bx + lx - 1;
        float v = FINF;                                     // OOB ignored by minpool
        if (gy >= 0 && gy < HH && gx >= 0 && gx < WW)
            v = (in[gy * WW + gx] > 0.5f) ? 1.0f : 0.0f;
        sM[i] = v;
    }
    __syncthreads();
    float cnt = 0.0f;
    for (int i = tid; i < TILE * TILE; i += 256) {
        int ly = i / TILE, lx = i % TILE;
        float ero = FINF;
        #pragma unroll
        for (int dy = 0; dy < 3; dy++)
            #pragma unroll
            for (int dx = 0; dx < 3; dx++)
                ero = fminf(ero, sM[(ly + dy) * 34 + (lx + dx)]);
        float m = sM[(ly + 1) * 34 + (lx + 1)];
        float e = ((ero > 0.5f) != (m > 0.5f)) ? 1.0f : 0.0f;
        out[(size_t)(by + ly) * WW + bx + lx] = e;
        cnt += e;
    }
    red[tid] = cnt;
    __syncthreads();
    for (int s = 128; s; s >>= 1) { if (tid < s) red[tid] += red[tid + s]; __syncthreads(); }
    if (tid == 0) atomicAdd(&acc[accIdx], red[0]);
}

// per-column fwd/bwd scans -> g^2   (which=0: ge->g2g, which=1: pe->g2p)
__global__ void edt_cols_kernel(const float* __restrict__ ge, const float* __restrict__ pe,
                                float* __restrict__ g2g, float* __restrict__ g2p) {
    int gid = blockIdx.x * blockDim.x + threadIdx.x;          // 0..8191
    int which = gid >> 12;
    int r = gid & 4095, sl = r >> 9, x = r & 511;
    const float* e = (which ? pe : ge)   + (size_t)sl * HH * WW + x;
    float*       g = (which ? g2p : g2g) + (size_t)sl * HH * WW + x;
    float d = 1024.0f;                                        // big = H+W
    for (int y = 0; y < HH; y++) {
        d = (e[(size_t)y * WW] > 0.5f) ? 0.0f : d + 1.0f;
        g[(size_t)y * WW] = d;
    }
    d = 1024.0f;
    for (int y = HH - 1; y >= 0; y--) {
        d = (e[(size_t)y * WW] > 0.5f) ? 0.0f : d + 1.0f;
        float m = fminf(d, g[(size_t)y * WW]);
        g[(size_t)y * WW] = m * m;
    }
}

// row lower-envelope + weighted accumulation; g2 row staged in LDS via async copy
__global__ void edt_rows_kernel(const float* __restrict__ g2p, const float* __restrict__ g2g,
                                const float* __restrict__ ge,  const float* __restrict__ pe,
                                float* __restrict__ acc) {
    const int y = blockIdx.x, sl = blockIdx.y, which = blockIdx.z;
    const float* g2 = which ? g2g : g2p;        // z=0: dist-to-pred-edges, weight ge
    const float* w  = which ? pe  : ge;
    const int accIdx = (which ? 16 : 8) + sl;
    const size_t rowOff = ((size_t)sl * HH + y) * WW;
    const int tid = threadIdx.x;

    __shared__ float rowS[WW];
    __shared__ float red[256];

    // 128 lanes x b128 = 2 KB row, one async burst (CDNA5 async-to-LDS path)
    if (tid < WW / 4) {
        unsigned ldsOff  = (unsigned)(unsigned long long)(&rowS[tid * 4]);
        const float* src = g2 + rowOff + tid * 4;
        asm volatile("global_load_async_to_lds_b128 %0, %1, off"
                     :: "v"(ldsOff), "v"(src) : "memory");
    }
    asm volatile("s_wait_asynccnt 0" ::: "memory");
    __syncthreads();

    float lsum = 0.0f;
    #pragma unroll
    for (int j = 0; j < WW / 256; j++) {
        int x = tid + j * 256;
        float best = 3.4e38f;
        for (int xp = 0; xp < WW; ++xp) {
            float dx = (float)(x - xp);
            best = fminf(best, fmaf(dx, dx, rowS[xp]));
        }
        lsum += sqrtf(best) * w[rowOff + x];
    }
    red[tid] = lsum;
    __syncthreads();
    for (int s = 128; s; s >>= 1) { if (tid < s) red[tid] += red[tid + s]; __syncthreads(); }
    if (tid == 0) atomicAdd(&acc[accIdx], red[0]);
}

// -------------------- finalize ---------------------------------------------
__global__ void finalize_kernel(const float* __restrict__ acc, float* __restrict__ out) {
    const float SM = 1e-6f;
    float tprec = (acc[0] + SM) / (acc[1] + SM);
    float tsens = (acc[2] + SM) / (acc[3] + SM);
    float cl = 1.0f - 2.0f * (tprec * tsens) / (tprec + tsens);
    float sum = 0.0f; int cnt = 0;
    for (int s = 0; s < 8; s++) {
        float g2pv = acc[8 + s]  / acc[24 + s];   // 0/0 -> nan (empty mean)
        float p2gv = acc[16 + s] / acc[32 + s];
        float ahd  = 0.5f * (g2pv + p2gv);
        float v    = 1.0f - 1.0f / (1.0f + ahd);
        if (!__builtin_isnan(v)) { sum += v; cnt++; }
    }
    float hd = cnt ? (sum / (float)cnt) : __builtin_nanf("");
    out[0] = cl;
    out[1] = hd;
    out[2] = 0.5f * cl + 0.5f * hd;                // ALPHA = 0.5
}

// ---------------------------------------------------------------------------
extern "C" void kernel_launch(void* const* d_in, const int* in_sizes, int n_in,
                              void* d_out, int out_size, void* d_ws, size_t ws_size,
                              hipStream_t stream) {
    const float* yT = (const float*)d_in[0];   // y_true
    const float* yP = (const float*)d_in[1];   // y_pred
    float* out = (float*)d_out;

    const size_t need = (size_t)6 * NPIX * sizeof(float) + 256 * sizeof(float);
    if (ws_size < need) return;                // avoid OOB if scratch undersized

    float* A0 = (float*)d_ws;
    float* A1 = A0 + (size_t)NPIX;
    float* A2 = A0 + (size_t)2 * NPIX;
    float* A3 = A0 + (size_t)3 * NPIX;
    float* A4 = A0 + (size_t)4 * NPIX;         // sk_pred
    float* A5 = A0 + (size_t)5 * NPIX;         // sk_true
    float* ACC = A0 + (size_t)6 * NPIX;

    dim3 blk(256);
    dim3 tgrid(WW / TILE, HH / TILE, 16);      // 16x16 tiles, 8 slices x 2 tensors

    zero_acc_kernel<<<1, 64, 0, stream>>>(ACC);

    // ---- clDice: soft skeletons (100 iterations), ping-pong im buffers ----
    skel_init_kernel<<<tgrid, blk, 0, stream>>>(yP, A4, yT, A5);
    const float* inP = yP;
    const float* inT = yT;
    for (int k = 0; k < 100; k++) {
        float* outP = (k & 1) ? A1 : A0;
        float* outT = (k & 1) ? A3 : A2;
        skel_iter_kernel<<<tgrid, blk, 0, stream>>>(inP, outP, A4, inT, outT, A5);
        inP = outP; inT = outT;
    }
    skel_reduce_kernel<<<32, 256, 0, stream>>>(A4, yT, ACC, 0);  // sum(skelP*yT), sum(skelP)
    skel_reduce_kernel<<<32, 256, 0, stream>>>(A5, yP, ACC, 2);  // sum(skelT*yP), sum(skelT)

    // ---- Hausdorff: edges -> column EDT -> row envelope + accumulation ----
    edges_kernel<<<tgrid, blk, 0, stream>>>(yT, yP, A0, A1, ACC);           // ge->A0, pe->A1
    edt_cols_kernel<<<32, 256, 0, stream>>>(A0, A1, A2, A3);                // g2g->A2, g2p->A3
    edt_rows_kernel<<<dim3(HH, NSL, 2), blk, 0, stream>>>(A3, A2, A0, A1, ACC);

    finalize_kernel<<<1, 1, 0, stream>>>(ACC, out);
}
